// MaskedSelfAttention_89635967468066
// MI455X (gfx1250) — matrix operand order repaired and checked by reading, not compiled, not run on hardware
//
#include <hip/hip_runtime.h>
#include <hip/hip_bf16.h>

// ---------------------------------------------------------------------------
// Masked causal self-attention for MI455X (gfx1250, wave32, WMMA).
//
// Compute-bound (~100 GFLOP vs ~100 MB @ 23.3 TB/s) -> all matmuls go through
// v_wmma_f32_16x16x32_bf16 (fp32 accumulate).  Operands are pre-laid-out in
// bf16 so every lane's fragment slice is a contiguous 16/32-byte global load
// (no LDS gather); the 192 MB L2 absorbs cross-tile re-reads.
//
// GEMM kernels: 16x64 per-wave tile (4 accumulators), 2-stage software
// pipeline (ping-pong fragment buffers, K-loop unrolled by 2), 256-thread
// blocks.  256 threads = 8 waves = 2 waves/SIMD resident minimum, which
// leaves a ~256-VGPR budget -- enough to keep BOTH pipeline stages hoisted
// (acc 32 + 2x32 B-frags + 2x8 A-frags + addressing ~ 140 VGPRs).  A
// 512-thread block capped the budget at ~128 and collapsed one stage back
// into load-then-consume stalls.
// ---------------------------------------------------------------------------

#define D_IN   1024
#define D_OUT  1024
#define SEQ    2048
#define BATCH  4
#define MTOT   (BATCH * SEQ)   // 8192 flattened (batch, seq) rows

typedef __attribute__((ext_vector_type(16))) __bf16 v16bf;
typedef __attribute__((ext_vector_type(8)))  float  v8f;

union Frag {
    unsigned int u[8];
    v16bf        v;
};

__device__ __forceinline__ unsigned short f2bf(float f) {
    unsigned int u = __float_as_uint(f);
    u += 0x7FFFu + ((u >> 16) & 1u);          // round-to-nearest-even
    return (unsigned short)(u >> 16);
}

__device__ __forceinline__ unsigned int pack2(float lo, float hi) {
    return (unsigned int)f2bf(lo) | ((unsigned int)f2bf(hi) << 16);
}

__device__ __forceinline__ v8f wmma_bf16(const Frag& a, const Frag& b, v8f c) {
    return __builtin_amdgcn_wmma_f32_16x16x32_bf16(
        false, a.v, false, b.v, (short)0, c, false, false);
}

// A-fragment (16x32 bf16), operand row-major in K.  Lane: row = lane&15,
// kg = lane>>4;  v0..3 = K kg*16..+7,  v4..7 = K kg*16+8..+15.
__device__ __forceinline__ void load_afrag(Frag& f, const unsigned short* rowptr,
                                           int kg) {
    const uint4 a = *(const uint4*)(rowptr + kg * 16);
    const uint4 b = *(const uint4*)(rowptr + kg * 16 + 8);
    f.u[0] = a.x; f.u[1] = a.y; f.u[2] = a.z; f.u[3] = a.w;
    f.u[4] = b.x; f.u[5] = b.y; f.u[6] = b.z; f.u[7] = b.w;
}

// B-fragment (32x16 bf16), operand stored K-contiguous per column.
// Lane: n = lane&15, K = (lane>>4)*16 .. +15  -> 32 contiguous bytes.
__device__ __forceinline__ void load_bfrag(Frag& f, const unsigned short* p) {
    const uint4 a = *(const uint4*)p;
    const uint4 b = *(const uint4*)(p + 8);
    f.u[0] = a.x; f.u[1] = a.y; f.u[2] = a.z; f.u[3] = a.w;
    f.u[4] = b.x; f.u[5] = b.y; f.u[6] = b.z; f.u[7] = b.w;
}

template <int NT>
__device__ __forceinline__ void load_btiles(Frag (&b)[NT],
                                            const unsigned short* base,
                                            size_t colstride16, int kb) {
#pragma unroll
    for (int t = 0; t < NT; t++)
        load_bfrag(b[t], base + (size_t)t * colstride16 + kb);
}

template <int NT>
__device__ __forceinline__ void gemm_step(const Frag& a, const Frag (&b)[NT],
                                          v8f (&acc)[NT]) {
#pragma unroll
    for (int t = 0; t < NT; t++) acc[t] = wmma_bf16(a, b[t], acc[t]);
}

// ---------------------------------------------------------------------------
// Kernel 0a: x (fp32) -> Xb (bf16).  8 elements / thread.
// ---------------------------------------------------------------------------
__global__ __launch_bounds__(256)
void convert_x_kernel(const float* __restrict__ x, unsigned short* __restrict__ Xb) {
    const size_t base = ((size_t)blockIdx.x * 256 + threadIdx.x) * 8;
    const float4 a = *(const float4*)(x + base);
    const float4 b = *(const float4*)(x + base + 4);
    uint4 o;
    o.x = pack2(a.x, a.y);
    o.y = pack2(a.z, a.w);
    o.z = pack2(b.x, b.y);
    o.w = pack2(b.z, b.w);
    *(uint4*)(Xb + base) = o;
}

// ---------------------------------------------------------------------------
// Kernel 0b: W[k][n] (fp32) -> Wt[n][k] (bf16), 32x32 LDS tile transpose.
// grid = (D/32, D/32, 3), block = 256.
// ---------------------------------------------------------------------------
__global__ __launch_bounds__(256)
void convert_wt_kernel(const float* __restrict__ Wq,
                       const float* __restrict__ Wk,
                       const float* __restrict__ Wv,
                       unsigned short* __restrict__ Wt) {
    __shared__ unsigned short tile[32][33];
    const float* W = (blockIdx.z == 0) ? Wq : (blockIdx.z == 1) ? Wk : Wv;
    unsigned short* dst = Wt + (size_t)blockIdx.z * D_IN * D_OUT;

    const int k0 = blockIdx.y * 32;            // K tile origin (rows of W)
    const int n0 = blockIdx.x * 32;            // N tile origin (cols of W)
    const int tid = threadIdx.x;

    {   // load 32x32 fp32 tile (coalesced), convert to bf16
        const int i = tid * 4;
        const int r = i >> 5, c = i & 31;
        const float4 v = *(const float4*)(W + (size_t)(k0 + r) * D_OUT + n0 + c);
        tile[r][c]     = f2bf(v.x);
        tile[r][c + 1] = f2bf(v.y);
        tile[r][c + 2] = f2bf(v.z);
        tile[r][c + 3] = f2bf(v.w);
    }
    __syncthreads();
    {   // write transposed: Wt[n][k], 4 contiguous k per thread
        const int i = tid * 4;
        const int n = i >> 5, kk = i & 31;
        unsigned short* o = dst + (size_t)(n0 + n) * D_IN + k0 + kk;
        o[0] = tile[kk][n];
        o[1] = tile[kk + 1][n];
        o[2] = tile[kk + 2][n];
        o[3] = tile[kk + 3][n];
    }
}

// ---------------------------------------------------------------------------
// Kernel 1: Q/K/V projection, LDS-free, double-buffered.
// grid = (MTOT/128, D_OUT/64, 3), block = 256 (8 waves x 16-row strips,
// 64-col block tile).  z==2 writes V transposed (Vt[b][d][s]).
// ---------------------------------------------------------------------------
__global__ __launch_bounds__(256)
void qkv_proj_kernel(const unsigned short* __restrict__ Xb,
                     const unsigned short* __restrict__ Wt,
                     const float* __restrict__ bq,
                     const float* __restrict__ bk,
                     const float* __restrict__ bv,
                     unsigned short* __restrict__ Qb,
                     unsigned short* __restrict__ Kb,
                     unsigned short* __restrict__ Vt) {
    const int z = blockIdx.z;
    const float* bias = (z == 0) ? bq : (z == 1) ? bk : bv;
    const unsigned short* W = Wt + (size_t)z * D_IN * D_OUT;

    const int tid  = threadIdx.x;
    const int wave = tid >> 5;
    const int lane = tid & 31;
    const int kg   = lane >> 4;
    const int ln   = lane & 15;
    const int m0   = blockIdx.x * 128;
    const int n0   = blockIdx.y * 64;

    v8f acc[4];
#pragma unroll
    for (int t = 0; t < 4; t++) acc[t] = {};

    const unsigned short* xrow  = Xb + (size_t)(m0 + wave * 16 + ln) * D_IN;
    const unsigned short* wbase = W + (size_t)(n0 + ln) * D_IN + kg * 16;
    const size_t cs16 = (size_t)16 * D_IN;

    Frag a0, a1;
    Frag b0[4], b1[4];
    load_afrag(a0, xrow, kg);
    load_btiles<4>(b0, wbase, cs16, 0);

    for (int kb = 0; kb < D_IN; kb += 64) {
        // prefetch stage 1 (kb+32) while stage 0 computes
        load_afrag(a1, xrow + kb + 32, kg);
        load_btiles<4>(b1, wbase, cs16, kb + 32);
        gemm_step<4>(a0, b0, acc);
        // prefetch stage 0 (kb+64) while stage 1 computes
        if (kb + 64 < D_IN) {
            load_afrag(a0, xrow + kb + 64, kg);
            load_btiles<4>(b0, wbase, cs16, kb + 64);
        }
        gemm_step<4>(a1, b1, acc);
    }

#pragma unroll
    for (int t = 0; t < 4; t++) {
        const int gn = n0 + t * 16 + ln;
        const float bv_ = bias[gn];
#pragma unroll
        for (int r = 0; r < 8; r++) {
            const int gm = m0 + wave * 16 + r + 8 * kg;   // = b*SEQ + s
            const unsigned short val = f2bf(acc[t][r] + bv_);
            if (z == 0) {
                Qb[(size_t)gm * D_OUT + gn] = val;
            } else if (z == 1) {
                Kb[(size_t)gm * D_OUT + gn] = val;
            } else {
                const int bb = gm >> 11;                  // / SEQ
                const int s  = gm & (SEQ - 1);
                Vt[((size_t)bb * D_OUT + gn) * SEQ + s] = val;
            }
        }
    }
}

// ---------------------------------------------------------------------------
// Kernel 2: scores + causal mask + softmax, double-buffered QK^T.
// grid = (SEQ/16, BATCH), block = 256, dynamic LDS = 16*SEQ*4 = 128 KB.
// ---------------------------------------------------------------------------
__global__ __launch_bounds__(256)
void scores_softmax_kernel(const unsigned short* __restrict__ Qb,
                           const unsigned short* __restrict__ Kb,
                           unsigned short* __restrict__ Pb) {
    extern __shared__ float S[];                // [16][SEQ] raw scores
    const int tid   = threadIdx.x;
    const int wave  = tid >> 5;
    const int lane  = tid & 31;
    const int kg    = lane >> 4;
    const int ln    = lane & 15;
    const int qt    = blockIdx.x;
    const int b     = blockIdx.y;
    const int qbase = qt * 16;
    const float scale = 0.03125f;               // 1/sqrt(1024)

    for (int i = tid; i < 16 * SEQ; i += 256) S[i] = -1e30f;
    __syncthreads();

    const unsigned short* qrow = Qb + (size_t)(b * SEQ + qbase + ln) * D_IN;
    const int ntiles = qt + 1;                  // causal: key tiles 0..qt
    for (int j = wave; j < ntiles; j += 8) {
        v8f acc = {};
        const unsigned short* krow =
            Kb + (size_t)(b * SEQ + j * 16 + ln) * D_IN + kg * 16;

        Frag a0, a1, f0, f1;
        load_afrag(a0, qrow, kg);
        load_bfrag(f0, krow);
        for (int kb = 0; kb < D_IN; kb += 64) {
            load_afrag(a1, qrow + kb + 32, kg);
            load_bfrag(f1, krow + kb + 32);
            acc = wmma_bf16(a0, f0, acc);
            if (kb + 64 < D_IN) {
                load_afrag(a0, qrow + kb + 64, kg);
                load_bfrag(f0, krow + kb + 64);
            }
            acc = wmma_bf16(a1, f1, acc);
        }
#pragma unroll
        for (int r = 0; r < 8; r++) {
            const int qr  = r + 8 * kg;         // local query row 0..15
            const int key = j * 16 + ln;
            S[qr * SEQ + key] = (key <= qbase + qr) ? acc[r] * scale : -1e30f;
        }
    }
    __syncthreads();

    // Row softmax: wave w owns rows 2w and 2w+1.
    for (int rr = wave * 2; rr < wave * 2 + 2; rr++) {
        float* row = &S[rr * SEQ];
        float m = -1e30f;
        for (int c = lane; c < SEQ; c += 32) m = fmaxf(m, row[c]);
#pragma unroll
        for (int o = 16; o > 0; o >>= 1) m = fmaxf(m, __shfl_xor(m, o, 32));
        float s = 0.0f;
        for (int c = lane; c < SEQ; c += 32) {
            const float e = __expf(row[c] - m);
            row[c] = e;
            s += e;
        }
#pragma unroll
        for (int o = 16; o > 0; o >>= 1) s += __shfl_xor(s, o, 32);
        const float inv = 1.0f / s;
        unsigned short* prow = Pb + (size_t)(b * SEQ + qbase + rr) * SEQ;
        for (int c = lane; c < SEQ; c += 32) prow[c] = f2bf(row[c] * inv);
    }
}

// ---------------------------------------------------------------------------
// Kernel 3: context = P @ V, LDS-free, double-buffered.
// grid = (SEQ/128, D_OUT/64, BATCH), block = 256 (8 waves, 16x64/wave).
// A-frags from row-major P, B-frags from transposed Vt.
// K-loop truncated at causal bound (P zero above diagonal).
// ---------------------------------------------------------------------------
__global__ __launch_bounds__(256)
void context_kernel(const unsigned short* __restrict__ Pb,
                    const unsigned short* __restrict__ Vt,
                    float* __restrict__ out) {
    const int tid  = threadIdx.x;
    const int wave = tid >> 5;
    const int lane = tid & 31;
    const int kg   = lane >> 4;
    const int ln   = lane & 15;
    const int m0   = blockIdx.x * 128;
    const int n0   = blockIdx.y * 64;
    const int b    = blockIdx.z;

    v8f acc[4];
#pragma unroll
    for (int t = 0; t < 4; t++) acc[t] = {};

    const unsigned short* prow =
        Pb + (size_t)(b * SEQ + m0 + wave * 16 + ln) * SEQ;
    const unsigned short* vbase =
        Vt + ((size_t)b * D_OUT + n0 + ln) * SEQ + kg * 16;
    const size_t cs16 = (size_t)16 * SEQ;

    const int kmax = m0 + 128;                  // keys beyond this have P == 0

    Frag a0, a1;
    Frag b0[4], b1[4];
    load_afrag(a0, prow, kg);
    load_btiles<4>(b0, vbase, cs16, 0);

    for (int kb = 0; kb < kmax; kb += 64) {
        load_afrag(a1, prow + kb + 32, kg);
        load_btiles<4>(b1, vbase, cs16, kb + 32);
        gemm_step<4>(a0, b0, acc);
        if (kb + 64 < kmax) {
            load_afrag(a0, prow + kb + 64, kg);
            load_btiles<4>(b0, vbase, cs16, kb + 64);
        }
        gemm_step<4>(a1, b1, acc);
    }

#pragma unroll
    for (int t = 0; t < 4; t++) {
        const int gn = n0 + t * 16 + ln;
#pragma unroll
        for (int r = 0; r < 8; r++) {
            const int gm = m0 + wave * 16 + r + 8 * kg;
            out[(size_t)(b * SEQ + gm) * D_OUT + gn] = acc[t][r];
        }
    }
}

// ---------------------------------------------------------------------------
// Host launch.  Workspace layout (bf16 = ushort elements):
//   Xb : MTOT*D_IN        (16 MB)
//   Wt : 3*D_IN*D_OUT     ( 6 MB)
//   Qb : MTOT*D_OUT       (16 MB)
//   Kb : MTOT*D_OUT       (16 MB)
//   Vt : MTOT*D_OUT       (16 MB, transposed [b][d][s])
//   Pb : BATCH*SEQ*SEQ    (32 MB)        total ~102 MB
// ---------------------------------------------------------------------------
extern "C" void kernel_launch(void* const* d_in, const int* in_sizes, int n_in,
                              void* d_out, int out_size, void* d_ws, size_t ws_size,
                              hipStream_t stream) {
    const float* x  = (const float*)d_in[0];
    const float* Wq = (const float*)d_in[1];
    const float* bq = (const float*)d_in[2];
    const float* Wk = (const float*)d_in[3];
    const float* bk = (const float*)d_in[4];
    const float* Wv = (const float*)d_in[5];
    const float* bv = (const float*)d_in[6];
    float* out = (float*)d_out;

    unsigned short* Xb = (unsigned short*)d_ws;
    unsigned short* Wt = Xb + (size_t)MTOT * D_IN;
    unsigned short* Qb = Wt + (size_t)3 * D_IN * D_OUT;
    unsigned short* Kb = Qb + (size_t)MTOT * D_OUT;
    unsigned short* Vt = Kb + (size_t)MTOT * D_OUT;
    unsigned short* Pb = Vt + (size_t)MTOT * D_OUT;

    convert_x_kernel<<<dim3((MTOT * D_IN) / (256 * 8)), dim3(256), 0, stream>>>(x, Xb);
    convert_wt_kernel<<<dim3(D_OUT / 32, D_IN / 32, 3), dim3(256), 0, stream>>>(
        Wq, Wk, Wv, Wt);

    qkv_proj_kernel<<<dim3(MTOT / 128, D_OUT / 64, 3), dim3(256), 0, stream>>>(
        Xb, Wt, bq, bk, bv, Qb, Kb, Vt);

    scores_softmax_kernel<<<dim3(SEQ / 16, BATCH), dim3(256),
                            16 * SEQ * sizeof(float), stream>>>(Qb, Kb, Pb);

    context_kernel<<<dim3(SEQ / 128, D_OUT / 64, BATCH), dim3(256), 0, stream>>>(
        Pb, Vt, out);
}
